// GraphConvLayer_22935125360693
// MI455X (gfx1250) — matrix operand order repaired and checked by reading, not compiled, not run on hardware
//
#include <hip/hip_runtime.h>
#include <hip/hip_bf16.h>

typedef __attribute__((ext_vector_type(16))) __bf16 v16bf;
typedef __attribute__((ext_vector_type(8)))  float  v8f;

#define IN_F   256
#define OUT_F  256
#define BOND_F 6
#define KCAT   262               // IN_F + BOND_F
#define KPAD   288               // KCAT rounded up to multiple of 32
#define LSTR   292               // LDS row stride (floats), padded vs 64 banks
#define BN_EPS 1e-5f

// ---------------------------------------------------------------------------
// Fused: self GEMM + gather-sum + degree GEMM + bias + ReLU -> total (d_out)
// block = 256 threads (8 wave32). Block tile: 16 rows x 128 cols (grid.y = 2).
// Each wave computes one 16x16 tile with v_wmma_f32_16x16x32_bf16.
// ---------------------------------------------------------------------------
__global__ __launch_bounds__(256) void graphconv_fused(
    const float* __restrict__ atom_features,   // [nAtoms, 256]
    const float* __restrict__ bond_features,   // [nBonds, 6]
    const int*   __restrict__ atom_nbrs,       // [n_d, deg]
    const int*   __restrict__ bond_nbrs,       // [n_d, deg]
    const float* __restrict__ W_self,          // [256, 256]
    const float* __restrict__ W_deg,           // [262, 256]
    const float* __restrict__ bias,            // [256]
    float*       __restrict__ out,             // [Ntotal, 256]
    int n_d, int deg, int rowOffset)
{
    __shared__ float lds[16 * LSTR];

    const int tid     = threadIdx.x;
    const int wave    = tid >> 5;             // 0..7
    const int lane    = tid & 31;
    const int tileRow = blockIdx.x * 16;      // local (degree-segment) row base
    const int colBase = blockIdx.y * 128 + wave * 16;

    // WMMA fragment addressing (wave32, 16-bit operands)
    const int mrow  = lane & 15;                   // A row / C col-lane
    const int aKoff = (lane >= 16) ? 8  : 0;       // A: hi half-wave K offset
    const int bKoff = (lane >= 16) ? 16 : 0;       // unused; B loads use aKoff too
    const int col   = colBase + (lane & 15);       // output column (0..255)

    // ---------------- Phase 1: stage 16x256 atom rows (self term) ----------
    {
        const int r    = tid >> 4;                 // 0..15
        const int c0   = tid & 15;
        const int arow = tileRow + r;
        if (arow < n_d) {
            const float* src = atom_features + (size_t)(rowOffset + arow) * IN_F;
            for (int c = c0; c < IN_F; c += 16) lds[r * LSTR + c] = src[c];
        } else {
            for (int c = c0; c < IN_F; c += 16) lds[r * LSTR + c] = 0.0f;
        }
    }
    __syncthreads();

    v8f acc = {};

    // self GEMM: K = 256, 8 WMMA steps
    for (int kb = 0; kb < IN_F; kb += 32) {
        v16bf af, bf;
#pragma unroll
        for (int e = 0; e < 16; ++e) {
            const int k = kb + e + aKoff + ((e >= 8) ? 8 : 0);
            af[e] = (__bf16)lds[mrow * LSTR + k];
        }
#pragma unroll
        for (int e = 0; e < 16; ++e) {
            const int k = kb + e + aKoff + ((e >= 8) ? 8 : 0);
            bf[e] = (__bf16)W_self[k * OUT_F + col];
        }
        acc = __builtin_amdgcn_wmma_f32_16x16x32_bf16(
                  false, af, false, bf, (short)0, acc, false, false);
    }
    __syncthreads();

    // ---------------- Phase 2: stage 16x288 gathered-sum rows --------------
    {
        const int r    = tid >> 4;
        const int c0   = tid & 15;
        const int arow = tileRow + r;
        // zero pad region [256, 288)
        for (int c = IN_F + c0; c < KPAD; c += 16) lds[r * LSTR + c] = 0.0f;
        if (arow < n_d) {
            for (int c = c0; c < IN_F; c += 16) {
                float s = 0.0f;
                for (int j = 0; j < deg; ++j) {
                    const int idx = atom_nbrs[arow * deg + j];
                    s += atom_features[(size_t)idx * IN_F + c];
                }
                lds[r * LSTR + c] = s;
            }
            if (c0 < BOND_F) {
                float s = 0.0f;
                for (int j = 0; j < deg; ++j) {
                    const int idx = bond_nbrs[arow * deg + j];
                    s += bond_features[(size_t)idx * BOND_F + c0];
                }
                lds[r * LSTR + IN_F + c0] = s;
            }
        } else {
            for (int c = c0; c < IN_F; c += 16) lds[r * LSTR + c] = 0.0f;
        }
    }
    __syncthreads();

    // degree GEMM: K = 262 (padded to 288), 9 WMMA steps
    for (int kb = 0; kb < KPAD; kb += 32) {
        v16bf af, bf;
#pragma unroll
        for (int e = 0; e < 16; ++e) {
            const int k = kb + e + aKoff + ((e >= 8) ? 8 : 0);
            af[e] = (__bf16)lds[mrow * LSTR + k];
        }
#pragma unroll
        for (int e = 0; e < 16; ++e) {
            const int k = kb + e + aKoff + ((e >= 8) ? 8 : 0);
            const float w = (k < KCAT) ? W_deg[k * OUT_F + col] : 0.0f;
            bf[e] = (__bf16)w;
        }
        acc = __builtin_amdgcn_wmma_f32_16x16x32_bf16(
                  false, af, false, bf, (short)0, acc, false, false);
    }

    // ---------------- Epilogue: bias + ReLU + store ------------------------
    const float b     = bias[col];
    const int   rbase = (lane >= 16) ? 8 : 0;
#pragma unroll
    for (int i = 0; i < 8; ++i) {
        const int arow = tileRow + rbase + i;
        if (arow < n_d) {
            float v = acc[i] + b;
            v = (v > 0.0f) ? v : 0.0f;
            out[(size_t)(rowOffset + arow) * OUT_F + col] = v;
        }
    }
}

// ---------------------------------------------------------------------------
// BN helper kernels
// ---------------------------------------------------------------------------
__global__ void zero_stats(float* __restrict__ stats) {
    if (threadIdx.x < 512) stats[threadIdx.x] = 0.0f;
}

__global__ __launch_bounds__(256) void col_stats(
    const float* __restrict__ t, float* __restrict__ stats, int N)
{
    const int col = threadIdx.x;                  // 256 threads = 256 columns
    const int rpb = (N + gridDim.x - 1) / (int)gridDim.x;
    const int r0  = blockIdx.x * rpb;
    int r1 = r0 + rpb; if (r1 > N) r1 = N;
    float s = 0.0f, q = 0.0f;
    for (int r = r0; r < r1; ++r) {
        const float v = t[(size_t)r * OUT_F + col];
        s += v; q += v * v;
    }
    atomicAdd(&stats[col], s);
    atomicAdd(&stats[OUT_F + col], q);
}

__global__ void finalize_stats(float* __restrict__ stats,
                               const float* __restrict__ bn_w,
                               const float* __restrict__ bn_b, int N)
{
    const int c = threadIdx.x;
    if (c < OUT_F) {
        const float invN = 1.0f / (float)N;
        const float mean = stats[c] * invN;
        float var = stats[OUT_F + c] * invN - mean * mean;
        if (var < 0.0f) var = 0.0f;
        const float inv   = rsqrtf(var + BN_EPS);
        const float scale = bn_w[c] * inv;
        stats[2 * OUT_F + c] = scale;
        stats[3 * OUT_F + c] = bn_b[c] - mean * scale;
    }
}

__global__ __launch_bounds__(256) void bn_apply(
    float* __restrict__ t, const float* __restrict__ stats, int N)
{
    const int col   = threadIdx.x;
    const float sc  = stats[2 * OUT_F + col];
    const float sh  = stats[3 * OUT_F + col];
    for (int r = blockIdx.x; r < N; r += gridDim.x) {
        const size_t idx = (size_t)r * OUT_F + col;
        t[idx] = t[idx] * sc + sh;
    }
}

// ---------------------------------------------------------------------------
// Launch
// ---------------------------------------------------------------------------
extern "C" void kernel_launch(void* const* d_in, const int* in_sizes, int n_in,
                              void* d_out, int out_size, void* d_ws, size_t ws_size,
                              hipStream_t stream) {
    const float* atom_features = (const float*)d_in[0];
    const float* bond_features = (const float*)d_in[1];

    const int*   atom_nbrs[5];
    const int*   bond_nbrs[5];
    int          nd[5];
    for (int d = 1; d <= 5; ++d) {
        atom_nbrs[d - 1] = (const int*)d_in[2 + 2 * (d - 1)];
        bond_nbrs[d - 1] = (const int*)d_in[3 + 2 * (d - 1)];
        nd[d - 1]        = in_sizes[2 + 2 * (d - 1)] / d;
    }
    const float* W_self = (const float*)d_in[12];
    const float* bias   = (const float*)d_in[13];
    const float* W_deg[5] = {
        (const float*)d_in[14], (const float*)d_in[15], (const float*)d_in[16],
        (const float*)d_in[17], (const float*)d_in[18]
    };
    const float* bn_w = (const float*)d_in[19];
    const float* bn_b = (const float*)d_in[20];

    float* out   = (float*)d_out;          // holds `total`, normalized in place
    float* stats = (float*)d_ws;           // [sum(256) | sumsq(256) | scale(256) | shift(256)]

    const int N = in_sizes[0] / IN_F;      // total atoms

    zero_stats<<<1, 512, 0, stream>>>(stats);

    int off = 0;
    for (int d = 0; d < 5; ++d) {
        const int n = nd[d];
        if (n > 0) {
            dim3 grid((n + 15) / 16, 2);
            graphconv_fused<<<grid, 256, 0, stream>>>(
                atom_features, bond_features, atom_nbrs[d], bond_nbrs[d],
                W_self, W_deg[d], bias, out, n, d + 1, off);
        }
        off += n;
    }

    col_stats<<<256, 256, 0, stream>>>(out, stats, N);
    finalize_stats<<<1, 256, 0, stream>>>(stats, bn_w, bn_b, N);
    bn_apply<<<1024, 256, 0, stream>>>(out, stats, N);
}
